// MSA_Block_68375879352483
// MI455X (gfx1250) — matrix operand order repaired and checked by reading, not compiled, not run on hardware
//
#include <hip/hip_runtime.h>

typedef unsigned short u16;
typedef unsigned int   u32;
typedef unsigned long long u64;
typedef __attribute__((ext_vector_type(16))) __bf16 v16bf;
typedef __attribute__((ext_vector_type(8)))  float  v8f;
typedef __attribute__((ext_vector_type(4)))  u32    u32x4;
typedef __attribute__((ext_vector_type(8)))  u32    u32x8;

union ABf { uint4 u[2]; v16bf v; };

// hardware RNE f32->bf16 (gfx1250 has native bf16 converts)
__device__ __forceinline__ u16 f2bf(float f) {
  __bf16 h = (__bf16)f;
  return __builtin_bit_cast(u16, h);
}

// A operand (16x32 bf16): lane&15 = row M; lane>=16 selects K+8 chunk.
// VGPRs 0..3 <- K base..base+7, VGPRs 4..7 <- K base+16..base+23 (ISA 7.12.2).
__device__ __forceinline__ v16bf loadA(const u16* base, int ld, int m0, int k0, int lane) {
  const int row = m0 + (lane & 15);
  const int kc  = k0 + ((lane >> 4) << 3);
  const u16* p = base + (size_t)row * ld + kc;
  ABf t;
  t.u[0] = *(const uint4*)(p);
  t.u[1] = *(const uint4*)(p + 16);
  return t.v;
}

// B operand (32x16 bf16): lane&15 = col N; lane>=16 selects K+8 chunk; elems 0..7 <- K base..+7, 8..15 <- K base+16..+23.
__device__ __forceinline__ v16bf loadB(const u16* base, int ld, int k0, int n0, int lane) {
  const u16* p = base + (size_t)(k0 + ((lane >> 4) << 3)) * ld + n0 + (lane & 15);
  u16 t[16];
#pragma unroll
  for (int s = 0; s < 8; ++s) {
    t[s]     = p[(size_t)s * ld];
    t[s + 8] = p[(size_t)(s + 16) * ld];
  }
  return __builtin_bit_cast(v16bf, t);
}

__device__ __forceinline__ v8f wmma_bf16(v16bf a, v16bf b, v8f c) {
  // (neg_a, A, neg_b, B, c_mod, C, reuse_a, reuse_b)
  return __builtin_amdgcn_wmma_f32_16x16x32_bf16(false, a, false, b, (short)0, c, false, false);
}

// ---- Tensor Data Mover helpers (CDNA5 ISA) ----
__device__ __forceinline__ unsigned lds_offset_of(const void* p) {
  // addrspace(3) pointers are 32-bit LDS byte offsets
  return (unsigned)(u64)(__attribute__((address_space(3))) const void*)p;
}

// 2D tile load: tensor = bf16 [512 x 512] row-major, tile = 32 rows x 256 cols.
// Groups 2/3 omitted (VADDR2/3 = NULL) -> 2D descriptor form.
__device__ __forceinline__ void tdm_load_wtile(const u16* gsrc, unsigned ldsOff) {
  const u64 ga = (u64)(size_t)gsrc;
  u32x4 g0;
  g0[0] = 1u;                                            // count=1, user descriptor
  g0[1] = ldsOff;                                        // lds_addr (bytes)
  g0[2] = (u32)ga;                                       // global_addr[31:0]
  g0[3] = (u32)((ga >> 32) & 0x01FFFFFFu) | (2u << 30);  // addr[56:32] | type=2
  u32x8 g1;
  g1[0] = 1u << 16;                                      // data_size = 2 bytes
  g1[1] = 512u << 16;                                    // tensor_dim0 = 512
  g1[2] = 512u << 16;                                    // tensor_dim1 = 512
  g1[3] = 256u << 16;                                    // tile_dim0 = 256
  g1[4] = 32u;                                           // tile_dim1 = 32, tile_dim2 = 0
  g1[5] = 512u;                                          // tensor_dim0_stride = 512
  g1[6] = 0u;
  g1[7] = 0u;
  asm volatile("tensor_load_to_lds %0, %1" :: "s"(g0), "s"(g1) : "memory");
}

// 64x64 = Xw[64,512](LDS bf16) @ W-slice[512,64](global bf16); bf16 result -> LDS
__device__ __forceinline__ void proj64(const u16* Xw, const u16* W, int lane,
                                       u16* dst, bool transpose) {
  v8f acc[4][4] = {};
#pragma unroll 1
  for (int kt = 0; kt < 16; ++kt) {
    const int k0 = kt * 32;
    v16bf A[4], B[4];
#pragma unroll
    for (int m = 0; m < 4; ++m) A[m] = loadA(Xw, 512, m * 16, k0, lane);
#pragma unroll
    for (int n = 0; n < 4; ++n) B[n] = loadB(W, 1536, k0, n * 16, lane);
#pragma unroll
    for (int m = 0; m < 4; ++m)
#pragma unroll
      for (int n = 0; n < 4; ++n) acc[m][n] = wmma_bf16(A[m], B[n], acc[m][n]);
  }
  const int jl = lane & 15, hl = lane >> 4;
#pragma unroll
  for (int m = 0; m < 4; ++m)
#pragma unroll
    for (int n = 0; n < 4; ++n)
#pragma unroll
      for (int r = 0; r < 8; ++r) {
        const int i = m * 16 + hl * 8 + r;   // C layout: M = vgpr + 8*(lane>=16)
        const int j = n * 16 + jl;           //           N = lane & 15
        const u16 h = f2bf(acc[m][n][r]);
        if (transpose) dst[j * 64 + i] = h;  // K^T for contiguous B loads later
        else           dst[i * 64 + j] = h;
      }
}

__global__ void cvt_bf16_kernel(const float* __restrict__ in, u16* __restrict__ o, int npair) {
  int i = blockIdx.x * blockDim.x + threadIdx.x;
  if (i < npair) {
    float2 f = ((const float2*)in)[i];
    ((u32*)o)[i] = (u32)f2bf(f.x) | ((u32)f2bf(f.y) << 16);
  }
}

// 1 WG per window (b*512 = 1024 WGs); 8 waves = 8 heads.
__global__ __launch_bounds__(256) void win_attn_kernel(
    const float* __restrict__ x, const u16* __restrict__ wqkv,
    const float* __restrict__ pos, u16* __restrict__ yout) {
  extern __shared__ char smem[];
  u16*   Xw   = (u16*)smem;                     // [64][512] bf16  (64 KB)
  u16*   wbuf = (u16*)smem + 32768;             // 8 waves * 16 KB ping-pong
  float* posl = (float*)(smem + 196608);        // [343]

  const int tid  = threadIdx.x;
  const int lane = tid & 31;
  const int wave = tid >> 5;                    // == head
  const int win  = blockIdx.x & 511;
  const int bat  = blockIdx.x >> 9;
  const int wl = win >> 6, wh = (win >> 3) & 7, ww = win & 7;

  for (int i = tid; i < 343; i += 256) posl[i] = pos[i];

  // stage x window -> LDS bf16 (coalesced f32 reads)
  for (int idx = tid; idx < 64 * 512; idx += 256) {
    const int t = idx >> 9, ch = idx & 511;
    const int a = t >> 4, bq = (t >> 2) & 3, c = t & 3;
    const size_t g = (size_t)(((bat * 32 + wl * 4 + a) * 32 + wh * 4 + bq) * 32 + ww * 4 + c);
    Xw[idx] = f2bf(x[g * 512 + ch]);
  }
  __syncthreads();

  u16* Qb = wbuf + wave * 8192;   // Q, later attn (bf16 [64][64])
  u16* Kt = Qb + 4096;            // K^T, later V

  proj64(Xw, wqkv + wave * 64,        lane, Qb, false);  // Q
  proj64(Xw, wqkv + 512  + wave * 64, lane, Kt, true);   // K^T

  // scores = Q @ K^T
  v8f acc[4][4] = {};
#pragma unroll
  for (int kt = 0; kt < 2; ++kt) {
    v16bf A[4], B[4];
#pragma unroll
    for (int m = 0; m < 4; ++m) A[m] = loadA(Qb, 64, m * 16, kt * 32, lane);
#pragma unroll
    for (int n = 0; n < 4; ++n) B[n] = loadB(Kt, 64, kt * 32, n * 16, lane);
#pragma unroll
    for (int m = 0; m < 4; ++m)
#pragma unroll
      for (int n = 0; n < 4; ++n) acc[m][n] = wmma_bf16(A[m], B[n], acc[m][n]);
  }

  // scale + relative-position bias + row softmax (rows live in 16-lane halves)
  const float scale = 0.125f;                   // 64^-0.5
  const int jl = lane & 15, hl = lane >> 4;
#pragma unroll
  for (int m = 0; m < 4; ++m) {
#pragma unroll
    for (int r = 0; r < 8; ++r) {
      const int i = m * 16 + hl * 8 + r;
      const int ai = i >> 4, bi = (i >> 2) & 3, ci = i & 3;
      float vals[4];
#pragma unroll
      for (int n = 0; n < 4; ++n) {
        const int j = n * 16 + jl;              // REL = coord_j - coord_i + 3
        const int idx = (((j >> 4) - ai + 3) * 7 + (((j >> 2) & 3) - bi + 3)) * 7
                        + ((j & 3) - ci + 3);
        vals[n] = acc[m][n][r] * scale + posl[idx];
      }
      float mx = fmaxf(fmaxf(vals[0], vals[1]), fmaxf(vals[2], vals[3]));
      for (int d = 1; d < 16; d <<= 1) mx = fmaxf(mx, __shfl_xor(mx, d, 32));
      float s = 0.f;
#pragma unroll
      for (int n = 0; n < 4; ++n) { vals[n] = __expf(vals[n] - mx); s += vals[n]; }
      for (int d = 1; d < 16; d <<= 1) s += __shfl_xor(s, d, 32);
      const float inv = 1.0f / s;
#pragma unroll
      for (int n = 0; n < 4; ++n) acc[m][n][r] = vals[n] * inv;
    }
  }

  // attn bf16 -> Qb (Q is dead)
#pragma unroll
  for (int m = 0; m < 4; ++m)
#pragma unroll
    for (int n = 0; n < 4; ++n)
#pragma unroll
      for (int r = 0; r < 8; ++r) {
        const int i = m * 16 + hl * 8 + r;
        Qb[i * 64 + n * 16 + jl] = f2bf(acc[m][n][r]);
      }

  proj64(Xw, wqkv + 1024 + wave * 64, lane, Kt, false);  // V over dead K^T

  // out = attn @ V
  v8f o[4][4] = {};
#pragma unroll
  for (int kt = 0; kt < 2; ++kt) {
    v16bf A[4], B[4];
#pragma unroll
    for (int m = 0; m < 4; ++m) A[m] = loadA(Qb, 64, m * 16, kt * 32, lane);
#pragma unroll
    for (int n = 0; n < 4; ++n) B[n] = loadB(Kt, 64, kt * 32, n * 16, lane);
#pragma unroll
    for (int m = 0; m < 4; ++m)
#pragma unroll
      for (int n = 0; n < 4; ++n) o[m][n] = wmma_bf16(A[m], B[n], o[m][n]);
  }

  // scatter to token-major [token, h*64+d] bf16 workspace
#pragma unroll
  for (int m = 0; m < 4; ++m)
#pragma unroll
    for (int r = 0; r < 8; ++r) {
      const int i = m * 16 + hl * 8 + r;
      const int a = i >> 4, bq = (i >> 2) & 3, c = i & 3;
      const size_t g = (size_t)(((bat * 32 + wl * 4 + a) * 32 + wh * 4 + bq) * 32 + ww * 4 + c);
      const size_t base = g * 512 + wave * 64 + jl;
#pragma unroll
      for (int n = 0; n < 4; ++n) yout[base + n * 16] = f2bf(o[m][n][r]);
    }
}

// [65536,512] bf16 @ [512,512] bf16 + bias -> f32.
// B operand tiles (32x256 of w_out) staged via Tensor Data Mover, double-buffered.
__global__ __launch_bounds__(256) void out_proj_kernel(
    const u16* __restrict__ y, const u16* __restrict__ wo,
    const float* __restrict__ bo, float* __restrict__ out) {
  __shared__ u16 wtile[2][32 * 256];            // 2 x 16 KB
  const int lane = threadIdx.x & 31;
  const int wave = threadIdx.x >> 5;
  const int row0 = blockIdx.x * 32 + (wave & 1) * 16;
  const int colB = blockIdx.y * 256;            // block column base
  const int colq = (wave >> 1) * 64;            // wave's quad within the 256 cols

  if (wave == 0)                                // prologue: tile for kt=0
    tdm_load_wtile(wo + colB, lds_offset_of(&wtile[0][0]));

  v8f acc[4] = {};
#pragma unroll 1
  for (int kt = 0; kt < 16; ++kt) {
    if (wave == 0) {
      if (kt + 1 < 16) {
        tdm_load_wtile(wo + (size_t)(kt + 1) * 32 * 512 + colB,
                       lds_offset_of(&wtile[(kt + 1) & 1][0]));
        __builtin_amdgcn_s_wait_tensorcnt(1);   // tile kt complete (in-order)
      } else {
        __builtin_amdgcn_s_wait_tensorcnt(0);
      }
    }
    __syncthreads();                            // tile kt visible to all waves
    const u16* wt = &wtile[kt & 1][0];
    const v16bf A = loadA(y + (size_t)row0 * 512, 512, 0, kt * 32, lane);
#pragma unroll
    for (int n = 0; n < 4; ++n) {
      const v16bf B = loadB(wt, 256, 0, colq + n * 16, lane);
      acc[n] = wmma_bf16(A, B, acc[n]);
    }
    __syncthreads();                            // all reads done before overwrite
  }
  const int jl = lane & 15, hl = lane >> 4;
#pragma unroll
  for (int n = 0; n < 4; ++n) {
    const int col = colB + colq + n * 16 + jl;
    const float bias = bo[col];
#pragma unroll
    for (int r = 0; r < 8; ++r) {
      const int row = row0 + hl * 8 + r;
      out[(size_t)row * 512 + col] = acc[n][r] + bias;
    }
  }
}

extern "C" void kernel_launch(void* const* d_in, const int* in_sizes, int n_in,
                              void* d_out, int out_size, void* d_ws, size_t ws_size,
                              hipStream_t stream) {
  const float* x      = (const float*)d_in[0];
  const float* wqkv_f = (const float*)d_in[1];
  const float* pos    = (const float*)d_in[2];
  const float* wout_f = (const float*)d_in[3];
  const float* bout   = (const float*)d_in[4];

  u16* ws_wqkv = (u16*)d_ws;                    // 512*1536 bf16
  u16* ws_wout = ws_wqkv + 512 * 1536;          // 512*512  bf16
  u16* ws_y    = ws_wout + 512 * 512;           // 65536*512 bf16 attn output

  cvt_bf16_kernel<<<(512 * 1536 / 2 + 255) / 256, 256, 0, stream>>>(wqkv_f, ws_wqkv, 512 * 1536 / 2);
  cvt_bf16_kernel<<<(512 * 512 / 2 + 255) / 256, 256, 0, stream>>>(wout_f, ws_wout, 512 * 512 / 2);

  const size_t smem = 196608 + 343 * sizeof(float) + 4;   // ~198 KB (<320 KB/WGP)
  win_attn_kernel<<<1024, 256, smem, stream>>>(x, ws_wqkv, pos, ws_y);

  out_proj_kernel<<<dim3(2048, 2), 256, 0, stream>>>(ws_y, ws_wout, bout, (float*)d_out);
}
